// DeformableConv2d_31224412242791
// MI455X (gfx1250) — matrix operand order repaired and checked by reading, not compiled, not run on hardware
//
#include <hip/hip_runtime.h>
#include <hip/hip_bf16.h>

typedef __attribute__((ext_vector_type(16))) _Float16 v16h;
typedef __attribute__((ext_vector_type(8)))  _Float16 v8h;
typedef __attribute__((ext_vector_type(8)))  float    v8f;

#define Bb    4
#define CINC  64
#define HH    96
#define WW    96
#define HW    (HH*WW)        // 9216
#define NPIX  (Bb*HW)        // 36864
#define KK9   9
#define CIN2  130
#define KDIM  (CINC*KK9)     // 576  (deform GEMM K)
#define KDIM2 (CIN2*KK9)     // 1170 (offset/mod GEMM K)
#define KPAD  1184           // 1170 padded to 37*32
#define COUT  64

// ---------- kernel 1a: w_reg f32 [64][64][3][3] -> f16 [64][576] (same flat order) ----------
__global__ void k_wcvt(const float* __restrict__ w, _Float16* __restrict__ wh, int n) {
    int i = blockIdx.x * 256 + threadIdx.x;
    if (i < n) wh[i] = (_Float16)w[i];
}

// ---------- kernel 1b: pack w_off (18 rows) + w_mod (9 rows) -> f16 [32][1184], zero-padded ----------
__global__ void k_wcat(const float* __restrict__ w_off, const float* __restrict__ w_mod,
                       _Float16* __restrict__ wcat) {
    int i = blockIdx.x * 256 + threadIdx.x;
    if (i >= 32 * KPAD) return;
    int oc = i / KPAD, K = i - oc * KPAD;
    float v = 0.f;
    if (K < KDIM2) {
        if (oc < 18)      v = w_off[oc * KDIM2 + K];          // [18][130*9], K = cin*9+tap
        else if (oc < 27) v = w_mod[(oc - 18) * KDIM2 + K];   // [9][130*9]
    }
    wcat[i] = (_Float16)v;
}

// ---------- kernel 2: offsets+modulators via WMMA ----------
// im2col(concat(fi,fj,fl)) [16 x 1184] x wcat^T [1184 x 32] per block.
// 64 threads = 2 wave32; wave w -> output cols 16w..16w+15 (27 of 32 cols used).
__global__ __launch_bounds__(64)
void k_offmod_wmma(const float* __restrict__ fi, const float* __restrict__ fj,
                   const float* __restrict__ fl,
                   const _Float16* __restrict__ wcat,
                   const float* __restrict__ b_off, const float* __restrict__ b_mod,
                   float* __restrict__ om) {
    __shared__ _Float16 ldsA[16][KPAD + 8];   // row stride 1192 halfs = 2384B (16B aligned)

    const int tid = threadIdx.x;
    const int P0  = blockIdx.x * 16;          // 16 pixels, same (b,h) since 16 | 96
    const int b   = P0 / HW;

    // ---- im2col fill (zero padding for borders and K/N pad) ----
    for (int i = tid; i < 16 * KPAD; i += 64) {
        int p = i / KPAD, K = i - p * KPAD;
        float v = 0.f;
        if (K < KDIM2) {
            int cin = K / 9, t = K - cin * 9;
            int dy = t / 3, dx = t - dy * 3;
            int pix = P0 + p;
            int rr = pix % HW, h = rr / WW, w = rr % WW;
            int y = h + dy - 1, x = w + dx - 1;
            if ((unsigned)y < (unsigned)HH && (unsigned)x < (unsigned)WW) {
                const float* src;
                if (cin < 64)       src = fi + (size_t)(b * 64 + cin) * HW;
                else if (cin < 128) src = fj + (size_t)(b * 64 + (cin - 64)) * HW;
                else                src = fl + (size_t)(b * 2  + (cin - 128)) * HW;
                v = src[y * WW + x];
            }
        }
        ldsA[p][K] = (_Float16)v;
    }
    __syncthreads();

    // ---- GEMM: 37 x v_wmma_f32_16x16x32_f16 per wave ----
    const int lane = tid & 31;
    const int wave = tid >> 5;
    const int hiK  = lane >> 4;
    const int arow = lane & 15;
    const int oc   = wave * 16 + (lane & 15);   // 0..31

    v8f acc = {0.f, 0.f, 0.f, 0.f, 0.f, 0.f, 0.f, 0.f};
    const _Float16* wrow = wcat + (size_t)oc * KPAD;

    for (int kb = 0; kb < KPAD; kb += 32) {
        v8h a0 = *(const v8h*)&ldsA[arow][kb + hiK * 16];
        v8h a1 = *(const v8h*)&ldsA[arow][kb + hiK * 16 + 8];
        v16h A;
        ((v8h*)&A)[0] = a0;
        ((v8h*)&A)[1] = a1;
        v16h Bt = *(const v16h*)&wrow[kb + hiK * 16];
        acc = __builtin_amdgcn_wmma_f32_16x16x32_f16(false, A, false, Bt,
                                                     (short)0, acc, false, false);
    }

    // ---- bias + 2*sigmoid epilogue; store om[pix][27] ----
    if (oc < 27) {
        float bias = (oc < 18) ? b_off[oc] : b_mod[oc - 18];
#pragma unroll
        for (int v = 0; v < 8; ++v) {
            int pix = P0 + v + 8 * hiK;
            float r = acc[v] + bias;
            if (oc >= 18) r = 2.f / (1.f + __expf(-r));
            om[(size_t)pix * 27 + oc] = r;
        }
    }
}

// ---------- kernel 3: deformable gather + WMMA GEMM ----------
// One block = 16 pixels x 64 out channels. 128 threads = 4 wave32, one 16x16 N-tile per wave.
__global__ __launch_bounds__(128)
void k_deform(const float* __restrict__ fj, const float* __restrict__ om,
              const _Float16* __restrict__ wh, float* __restrict__ out) {
    __shared__ _Float16 ldsA[16][KDIM + 8];  // row stride 584 halfs = 1168B (16B aligned)
    __shared__ int   idx4[144][4];
    __shared__ float wt4[144][4];

    const int tid = threadIdx.x;
    const int P0  = blockIdx.x * 16;
    const int b   = P0 / HW;

    // ---- phase 1: per (pixel,tap) sampling coords + premultiplied bilinear weights ----
    for (int j = tid; j < 144; j += 128) {
        int p = j / 9, k = j - p * 9;
        int pix = P0 + p;
        int rr = pix % HW, h = rr / WW, w = rr % WW;
        const float* o = om + (size_t)pix * 27;
        float dy = o[2 * k], dx = o[2 * k + 1], m = o[18 + k];
        int ky = k / 3, kx = k - ky * 3;
        float py = (float)(h - 1 + ky) + dy;
        float px = (float)(w - 1 + kx) + dx;
        float y0f = floorf(py), x0f = floorf(px);
        float wy = py - y0f, wx = px - x0f;
        int y0 = (int)y0f, x0 = (int)x0f;
#pragma unroll
        for (int cy = 0; cy < 2; ++cy) {
#pragma unroll
            for (int cx = 0; cx < 2; ++cx) {
                int yc = y0 + cy, xc = x0 + cx;
                bool valid = (yc >= 0) & (yc < HH) & (xc >= 0) & (xc < WW);
                float wgt = (cy ? wy : 1.f - wy) * (cx ? wx : 1.f - wx) * m;
                if (!valid) wgt = 0.f;
                int yci = min(max(yc, 0), HH - 1);
                int xci = min(max(xc, 0), WW - 1);
                wt4[j][cy * 2 + cx]  = wgt;
                idx4[j][cy * 2 + cx] = yci * WW + xci;
            }
        }
    }
    __syncthreads();

    // ---- phase 2: gather into A tile [p][K = c*9 + k] as f16 ----
    const float* fjb = fj + (size_t)b * CINC * HW;
    for (int i = tid; i < 16 * KDIM; i += 128) {
        int p = i / KDIM, K = i - p * KDIM;
        int c = K / 9, k = K - c * 9;
        int j = p * 9 + k;
        const float* src = fjb + (size_t)c * HW;
        float v = wt4[j][0] * src[idx4[j][0]] + wt4[j][1] * src[idx4[j][1]]
                + wt4[j][2] * src[idx4[j][2]] + wt4[j][3] * src[idx4[j][3]];
        ldsA[p][K] = (_Float16)v;
    }
    __syncthreads();

    // ---- phase 3: [16 x 576] x [576 x 16] per wave via v_wmma_f32_16x16x32_f16 ----
    const int lane = tid & 31;
    const int wave = tid >> 5;
    const int hiK  = lane >> 4;
    const int arow = lane & 15;
    const int oc   = wave * 16 + (lane & 15);

    v8f acc = {0.f, 0.f, 0.f, 0.f, 0.f, 0.f, 0.f, 0.f};
    const _Float16* wrow = wh + (size_t)oc * KDIM;

    for (int kb = 0; kb < KDIM; kb += 32) {
        v8h a0 = *(const v8h*)&ldsA[arow][kb + hiK * 16];
        v8h a1 = *(const v8h*)&ldsA[arow][kb + hiK * 16 + 8];
        v16h A;
        ((v8h*)&A)[0] = a0;
        ((v8h*)&A)[1] = a1;
        v16h Bt = *(const v16h*)&wrow[kb + hiK * 16];
        acc = __builtin_amdgcn_wmma_f32_16x16x32_f16(false, A, false, Bt,
                                                     (short)0, acc, false, false);
    }

#pragma unroll
    for (int v = 0; v < 8; ++v) {
        int pix = P0 + v + 8 * hiK;
        int hw  = pix % HW;
        out[((size_t)(b * COUT) + oc) * HW + hw] = acc[v];
    }
}

extern "C" void kernel_launch(void* const* d_in, const int* in_sizes, int n_in,
                              void* d_out, int out_size, void* d_ws, size_t ws_size,
                              hipStream_t stream) {
    const float* fi    = (const float*)d_in[0];
    const float* fj    = (const float*)d_in[1];
    const float* fl    = (const float*)d_in[2];
    const float* w_off = (const float*)d_in[3];
    const float* b_off = (const float*)d_in[4];
    const float* w_mod = (const float*)d_in[5];
    const float* b_mod = (const float*)d_in[6];
    const float* w_reg = (const float*)d_in[7];
    float* out = (float*)d_out;

    float*    om   = (float*)d_ws;                                      // [NPIX][27] f32
    _Float16* wh   = (_Float16*)((char*)d_ws + (size_t)NPIX * 27 * 4);  // [64][576]  f16
    _Float16* wcat = wh + (size_t)COUT * KDIM;                          // [32][1184] f16 (32B-aligned)

    k_wcvt<<<(COUT * KDIM + 255) / 256, 256, 0, stream>>>(w_reg, wh, COUT * KDIM);
    k_wcat<<<(32 * KPAD + 255) / 256,   256, 0, stream>>>(w_off, w_mod, wcat);
    k_offmod_wmma<<<NPIX / 16, 64,  0, stream>>>(fi, fj, fl, wcat, b_off, b_mod, om);
    k_deform     <<<NPIX / 16, 128, 0, stream>>>(fj, om, wh, out);
}
